// LinearAttention_36197984371273
// MI455X (gfx1250) — matrix operand order repaired and checked by reading, not compiled, not run on hardware
//
#include <hip/hip_runtime.h>
#include <hip/hip_bf16.h>

// ---------------------------------------------------------------------------
// LinearAttention on gfx1250 (MI455X): bf16 WMMA GEMMs + fused softmax / LN.
// B=16, C=256, HEADS=8, d=32, HW=4096. All GEMMs use v_wmma_f32_16x16x32_bf16.
// All fragment gathers are 128-bit; all staging is static + vectorized.
// ---------------------------------------------------------------------------

typedef __attribute__((ext_vector_type(16))) __bf16 v16bf;
typedef __attribute__((ext_vector_type(8)))  __bf16 v8bf;
typedef __attribute__((ext_vector_type(4)))  __bf16 v4bf;
typedef __attribute__((ext_vector_type(8)))  float  v8f;
typedef __attribute__((ext_vector_type(4)))  float  v4f;
typedef __bf16 bf16_t;

#define NB     16      // batches
#define CDIM   256     // channels
#define HEADS  8
#define DHEAD  32
#define HWN    4096    // 64*64 spatial
#define QKVC   768     // 3*CDIM

__device__ __forceinline__ v16bf cat8(v8bf lo, v8bf hi) {
    return __builtin_shufflevector(lo, hi, 0, 1, 2, 3, 4, 5, 6, 7,
                                           8, 9, 10, 11, 12, 13, 14, 15);
}

__device__ __forceinline__ v8f wmma_bf16(v16bf a, v16bf b, v8f c) {
    return __builtin_amdgcn_wmma_f32_16x16x32_bf16(
        false, a, false, b, (short)0, c, false, false);
}

// ---------------------------------------------------------------------------
// Kernel 1: QKV = W_qkv(768x256) x X_b(256x4096), fp32 out.
// Block = 128 thr (4 waves). Block tile M=16, N=128, KSTEP=64.
// Wave = two 16x16 D tiles sharing one A fragment; 4 WMMAs per k-step.
// LDS: A row-major 16x64; B column-major 128x64 ([n][k]) so fragment gathers
// are ds_load_b128. Staging: A = 2x global b128 + 1x ds_store_b128/thread;
// B = coalesced 32B row chunks transposed in-reg, ds_store_b64 k-packs.
// ---------------------------------------------------------------------------
__global__ __launch_bounds__(128) void qkv_gemm_kernel(
    const float* __restrict__ x, const float* __restrict__ w_qkv,
    float* __restrict__ qkv)
{
    __shared__ bf16_t lA[16 * 64];     // [m][k]
    __shared__ bf16_t lBc[128 * 64];   // [n][k]

    const int n0 = blockIdx.x * 128;
    const int m0 = blockIdx.y * 16;
    const int b  = blockIdx.z;

    const int tid  = threadIdx.x;
    const int wv   = tid >> 5;
    const int lane = tid & 31;
    const int m    = lane & 15;
    const int hf   = lane >> 4;
    const int col  = lane & 15;

    const long xbase = (long)b * CDIM * HWN;
    const long qbase = (long)b * QKVC * HWN;

    const int rA  = tid >> 3;          // A stage: row 0..15
    const int cA  = (tid & 7) * 8;     // A stage: col chunk
    const int csB = (tid & 15) * 8;    // B stage: col start (N)
    const int r8  = (tid >> 4) * 8;    // B stage: row start (K)

    v8f acc0 = {}, acc1 = {};

    for (int k0 = 0; k0 < CDIM; k0 += 64) {
        // ---- stage A (16x64 of W): static, wide
        {
            const float* src = w_qkv + (m0 + rA) * CDIM + k0 + cA;
            v4f f0 = *(const v4f*)(src);
            v4f f1 = *(const v4f*)(src + 4);
            v8bf o;
#pragma unroll
            for (int j = 0; j < 4; ++j) { o[j] = (bf16_t)f0[j]; o[j + 4] = (bf16_t)f1[j]; }
            *(v8bf*)&lA[rA * 64 + cA] = o;
        }
        // ---- stage B (64x128 of X) transposed into [n][k], 8x8 per thread
#pragma unroll
        for (int g = 0; g < 8; g += 4) {
            v4f f[4][2];
#pragma unroll
            for (int j = 0; j < 4; ++j) {
                const float* src = x + xbase + (long)(k0 + r8 + g + j) * HWN + n0 + csB;
                f[j][0] = *(const v4f*)(src);
                f[j][1] = *(const v4f*)(src + 4);
            }
#pragma unroll
            for (int c = 0; c < 8; ++c) {
                v4bf pk;
#pragma unroll
                for (int j = 0; j < 4; ++j) pk[j] = (bf16_t)f[j][c >> 2][c & 3];
                *(v4bf*)&lBc[(csB + c) * 64 + r8 + g] = pk;   // ds_store_b64
            }
        }
        __syncthreads();

#pragma unroll
        for (int kk = 0; kk < 64; kk += 32) {
            v16bf a = cat8(*(const v8bf*)&lA[m * 64 + kk + 16 * hf],
                           *(const v8bf*)&lA[m * 64 + kk + 16 * hf + 8]);
            const int n = wv * 32 + col;
            v16bf b0 = cat8(*(const v8bf*)&lBc[n * 64 + kk + 16 * hf],
                            *(const v8bf*)&lBc[n * 64 + kk + 16 * hf + 8]);
            acc0 = wmma_bf16(a, b0, acc0);
            v16bf b1 = cat8(*(const v8bf*)&lBc[(n + 16) * 64 + kk + 16 * hf],
                            *(const v8bf*)&lBc[(n + 16) * 64 + kk + 16 * hf + 8]);
            acc1 = wmma_bf16(a, b1, acc1);
        }
        __syncthreads();
    }

    // D layout: VGPR r -> row (r + 8*hf), lanes 0..15 -> columns (coalesced)
#pragma unroll
    for (int r = 0; r < 8; ++r) {
        const long row = qbase + (long)(m0 + r + 8 * hf) * HWN + n0 + wv * 32;
        qkv[row + col]      = acc0[r];
        qkv[row + 16 + col] = acc1[r];
    }
}

// ---------------------------------------------------------------------------
// Kernel 2: softmax over head-dim d (32 strided rows) for Q;
// writes q position-major bf16: qT[(b*8+h)*4096 + n][e]  (64B vector stores)
// ---------------------------------------------------------------------------
__global__ __launch_bounds__(256) void q_softmax_kernel(
    const float* __restrict__ qkv, bf16_t* __restrict__ qT)
{
    const int n = blockIdx.x * 256 + threadIdx.x;
    const int h = blockIdx.y;
    const int b = blockIdx.z;
    const long base = (long)b * QKVC * HWN + (long)(h * DHEAD) * HWN + n;

    float v[DHEAD];
    float mx = -3.0e38f;
#pragma unroll
    for (int d = 0; d < DHEAD; ++d) {
        v[d] = qkv[base + (long)d * HWN];
        mx = fmaxf(mx, v[d]);
    }
    float s = 0.f;
#pragma unroll
    for (int d = 0; d < DHEAD; ++d) { v[d] = __expf(v[d] - mx); s += v[d]; }
    const float inv = 1.f / s;
    bf16_t* o = qT + ((long)(b * HEADS + h) * HWN + n) * DHEAD;
#pragma unroll
    for (int g = 0; g < 4; ++g) {
        v8bf o8;
#pragma unroll
        for (int j = 0; j < 8; ++j) o8[j] = (bf16_t)(v[g * 8 + j] * inv);
        *(v8bf*)(o + g * 8) = o8;
    }
}

// ---------------------------------------------------------------------------
// Kernel 3: softmax over y (64 contiguous) for K; bf16 out [B,256,4096]
// ---------------------------------------------------------------------------
__global__ __launch_bounds__(256) void k_softmax_kernel(
    const float* __restrict__ qkv, bf16_t* __restrict__ kS)
{
    const int idx = blockIdx.x * 256 + threadIdx.x;   // over 16*256*64
    const int b  = idx >> 14;
    const int rc = idx & 16383;
    const int c  = rc >> 6;     // channel within k-block (head*32 + d)
    const int xr = rc & 63;     // spatial row

    const float* p = qkv + (long)b * QKVC * HWN + (long)(CDIM + c) * HWN + xr * 64;
    float v[64];
#pragma unroll
    for (int g = 0; g < 16; ++g) {
        v4f f = *(const v4f*)(p + g * 4);
#pragma unroll
        for (int j = 0; j < 4; ++j) v[g * 4 + j] = f[j];
    }
    float mx = -3.0e38f;
#pragma unroll
    for (int y = 0; y < 64; ++y) mx = fmaxf(mx, v[y]);
    float s = 0.f;
#pragma unroll
    for (int y = 0; y < 64; ++y) { v[y] = __expf(v[y] - mx); s += v[y]; }
    const float inv = 1.f / s;
    bf16_t* o = kS + (long)b * CDIM * HWN + (long)c * HWN + xr * 64;
#pragma unroll
    for (int g = 0; g < 8; ++g) {
        v8bf o8;
#pragma unroll
        for (int j = 0; j < 8; ++j) o8[j] = (bf16_t)(v[g * 8 + j] * inv);
        *(v8bf*)(o + g * 8) = o8;
    }
}

// ---------------------------------------------------------------------------
// Kernel 4: V fp32 -> bf16 [B,256,4096] (8 elems/thread, b128 traffic)
// ---------------------------------------------------------------------------
__global__ __launch_bounds__(256) void v_convert_kernel(
    const float* __restrict__ qkv, bf16_t* __restrict__ vB)
{
    const long i0 = ((long)blockIdx.x * 256 + threadIdx.x) * 8;
    const long per = (long)CDIM * HWN;
    const long b   = i0 / per;
    const long r   = i0 - b * per;
    const float* p = qkv + b * QKVC * HWN + (long)(2 * CDIM) * HWN + r;
    v4f f0 = *(const v4f*)(p);
    v4f f1 = *(const v4f*)(p + 4);
    v8bf o;
#pragma unroll
    for (int j = 0; j < 4; ++j) { o[j] = (bf16_t)f0[j]; o[j + 4] = (bf16_t)f1[j]; }
    *(v8bf*)(vB + i0) = o;
}

// ---------------------------------------------------------------------------
// Kernel 5: context partials. ctx[b,h](32x32) = kS(32x4096) x vB^T(4096x32),
// K split into 8 chunks of 512 for occupancy; deterministic partial buffer.
// Both operands contiguous along K -> global_load_b128 fragment gathers.
// ---------------------------------------------------------------------------
__global__ __launch_bounds__(128) void context_part_kernel(
    const bf16_t* __restrict__ kS, const bf16_t* __restrict__ vB,
    float* __restrict__ part)
{
    const int bh   = blockIdx.x;          // 0..127
    const int kc   = blockIdx.y;          // 0..7 K-chunk
    const int b    = bh >> 3;
    const int head = bh & 7;
    const int wv   = threadIdx.x >> 5;
    const int lane = threadIdx.x & 31;
    const int m    = lane & 15;
    const int hf   = lane >> 4;
    const int col  = lane & 15;
    const int m0   = (wv >> 1) * 16;      // d-tile
    const int n0   = (wv & 1) * 16;       // e-tile

    const long kb = ((long)b * CDIM + head * DHEAD) * HWN;
    const bf16_t* arow = kS + kb + (long)(m0 + m) * HWN;
    const bf16_t* brow = vB + kb + (long)(n0 + col) * HWN;

    v8f acc = {};
    const int kbeg = kc * 512, kend = kbeg + 512;
#pragma unroll 4
    for (int k0 = kbeg; k0 < kend; k0 += 32) {
        v16bf a  = cat8(*(const v8bf*)(arow + k0 + 16 * hf),
                        *(const v8bf*)(arow + k0 + 16 * hf + 8));
        v16bf bb = cat8(*(const v8bf*)(brow + k0 + 16 * hf),
                        *(const v8bf*)(brow + k0 + 16 * hf + 8));
        acc = wmma_bf16(a, bb, acc);
    }
#pragma unroll
    for (int r = 0; r < 8; ++r)
        part[((long)kc * 128 + bh) * 1024 + (m0 + r + 8 * hf) * 32 + n0 + col] = acc[r];
}

// ---------------------------------------------------------------------------
// Kernel 6: reduce the 8 K-chunk partials (deterministic, no atomics)
// ---------------------------------------------------------------------------
__global__ __launch_bounds__(256) void ctx_reduce_kernel(
    const float* __restrict__ part, float* __restrict__ ctx)
{
    const int i = blockIdx.x * 256 + threadIdx.x;   // over 128*1024
    float s = 0.f;
#pragma unroll
    for (int kc = 0; kc < 8; ++kc) s += part[(long)kc * 131072 + i];
    ctx[i] = s;
}

// ---------------------------------------------------------------------------
// Kernel 7: attn[b,h](32x4096) = ctx(32x32) x qT(32x4096). One WMMA per tile
// (K=32 == head dim); wave does two n-tiles reusing its A fragment.
// qT is position-major -> B columns are two global_load_b128 per lane.
// ---------------------------------------------------------------------------
__global__ __launch_bounds__(128) void apply_attn_kernel(
    const float* __restrict__ ctx, const bf16_t* __restrict__ qT,
    bf16_t* __restrict__ attn)
{
    const int wv   = threadIdx.x >> 5;
    const int lane = threadIdx.x & 31;
    const int m    = lane & 15;
    const int hf   = lane >> 4;
    const int col  = lane & 15;
    const int nb   = blockIdx.x * 128 + wv * 32;
    const int m0   = blockIdx.y * 16;     // 0 or 16 within the head
    const int bh   = blockIdx.z;
    const int b    = bh >> 3;
    const int head = bh & 7;

    // A = context rows (m0..m0+15) x K=32, fp32 -> bf16 in-reg (vector loads)
    const float* cr = ctx + (long)bh * 1024 + (long)(m0 + m) * 32;
    v4f c0 = *(const v4f*)(cr + 16 * hf);
    v4f c1 = *(const v4f*)(cr + 16 * hf + 4);
    v4f c2 = *(const v4f*)(cr + 16 * hf + 8);
    v4f c3 = *(const v4f*)(cr + 16 * hf + 12);
    v16bf a;
#pragma unroll
    for (int i = 0; i < 4; ++i) {
        a[i]      = (bf16_t)c0[i];
        a[i + 4]  = (bf16_t)c1[i];
        a[i + 8]  = (bf16_t)c2[i];
        a[i + 12] = (bf16_t)c3[i];
    }

    v8f acc0 = {}, acc1 = {};
    {
        const bf16_t* qr = qT + ((long)bh * HWN + nb + col) * DHEAD + 16 * hf;
        v16bf bb = cat8(*(const v8bf*)qr, *(const v8bf*)(qr + 8));
        acc0 = wmma_bf16(a, bb, acc0);
    }
    {
        const bf16_t* qr = qT + ((long)bh * HWN + nb + 16 + col) * DHEAD + 16 * hf;
        v16bf bb = cat8(*(const v8bf*)qr, *(const v8bf*)(qr + 8));
        acc1 = wmma_bf16(a, bb, acc1);
    }

    const long obase = ((long)b * CDIM + head * DHEAD + m0) * HWN;
#pragma unroll
    for (int r = 0; r < 8; ++r) {
        const long row = obase + (long)(r + 8 * hf) * HWN + nb;
        attn[row + col]      = (bf16_t)acc0[r];
        attn[row + 16 + col] = (bf16_t)acc1[r];
    }
}

// ---------------------------------------------------------------------------
// Kernel 8: Y0 = W_out(256x256) x attn(256x4096) + b_out, fp32 out.
// Same KSTEP=64 LDS-staged skeleton as kernel 1 (B source already bf16).
// ---------------------------------------------------------------------------
__global__ __launch_bounds__(128) void out_proj_kernel(
    const bf16_t* __restrict__ attn, const float* __restrict__ w_out,
    const float* __restrict__ b_out, float* __restrict__ y0)
{
    __shared__ bf16_t lA[16 * 64];
    __shared__ bf16_t lBc[128 * 64];

    const int n0 = blockIdx.x * 128;
    const int m0 = blockIdx.y * 16;
    const int b  = blockIdx.z;
    const int tid  = threadIdx.x;
    const int wv   = tid >> 5;
    const int lane = tid & 31;
    const int m    = lane & 15;
    const int hf   = lane >> 4;
    const int col  = lane & 15;
    const int rA  = tid >> 3;
    const int cA  = (tid & 7) * 8;
    const int csB = (tid & 15) * 8;
    const int r8  = (tid >> 4) * 8;

    const long abase = (long)b * CDIM * HWN;
    v8f acc0 = {}, acc1 = {};

    for (int k0 = 0; k0 < CDIM; k0 += 64) {
        {
            const float* src = w_out + (m0 + rA) * CDIM + k0 + cA;
            v4f f0 = *(const v4f*)(src);
            v4f f1 = *(const v4f*)(src + 4);
            v8bf o;
#pragma unroll
            for (int j = 0; j < 4; ++j) { o[j] = (bf16_t)f0[j]; o[j + 4] = (bf16_t)f1[j]; }
            *(v8bf*)&lA[rA * 64 + cA] = o;
        }
#pragma unroll
        for (int g = 0; g < 8; g += 4) {
            v8bf f[4];
#pragma unroll
            for (int j = 0; j < 4; ++j)
                f[j] = *(const v8bf*)(attn + abase + (long)(k0 + r8 + g + j) * HWN + n0 + csB);
#pragma unroll
            for (int c = 0; c < 8; ++c) {
                v4bf pk;
#pragma unroll
                for (int j = 0; j < 4; ++j) pk[j] = f[j][c];
                *(v4bf*)&lBc[(csB + c) * 64 + r8 + g] = pk;
            }
        }
        __syncthreads();

#pragma unroll
        for (int kk = 0; kk < 64; kk += 32) {
            v16bf a = cat8(*(const v8bf*)&lA[m * 64 + kk + 16 * hf],
                           *(const v8bf*)&lA[m * 64 + kk + 16 * hf + 8]);
            const int n = wv * 32 + col;
            v16bf b0 = cat8(*(const v8bf*)&lBc[n * 64 + kk + 16 * hf],
                            *(const v8bf*)&lBc[n * 64 + kk + 16 * hf + 8]);
            acc0 = wmma_bf16(a, b0, acc0);
            v16bf b1 = cat8(*(const v8bf*)&lBc[(n + 16) * 64 + kk + 16 * hf],
                            *(const v8bf*)&lBc[(n + 16) * 64 + kk + 16 * hf + 8]);
            acc1 = wmma_bf16(a, b1, acc1);
        }
        __syncthreads();
    }

#pragma unroll
    for (int r = 0; r < 8; ++r) {
        const int row = m0 + r + 8 * hf;
        const float bias = b_out[row];
        const long base = (long)b * CDIM * HWN + (long)row * HWN + n0 + wv * 32;
        y0[base + col]      = acc0[r] + bias;
        y0[base + 16 + col] = acc1[r] + bias;
    }
}

// ---------------------------------------------------------------------------
// Kernel 9: LayerNorm over channels (256) per spatial position.
// Thread t = position n: for fixed c, adjacent threads -> adjacent addresses.
// ---------------------------------------------------------------------------
__global__ __launch_bounds__(256) void layernorm_kernel(
    const float* __restrict__ y0, const float* __restrict__ gamma,
    const float* __restrict__ beta, float* __restrict__ out)
{
    const int n = blockIdx.x * 256 + threadIdx.x;
    const int b = blockIdx.y;
    const float* p = y0 + (long)b * CDIM * HWN + n;

    float s = 0.f, ss = 0.f;
#pragma unroll 8
    for (int c = 0; c < CDIM; ++c) {
        float v = p[(long)c * HWN];
        s += v; ss += v * v;
    }
    const float mean = s * (1.f / CDIM);
    const float var  = ss * (1.f / CDIM) - mean * mean;
    const float rstd = rsqrtf(var + 1e-5f);

    float* o = out + (long)b * CDIM * HWN + n;
#pragma unroll 8
    for (int c = 0; c < CDIM; ++c) {
        float v = p[(long)c * HWN];
        o[(long)c * HWN] = (v - mean) * rstd * gamma[c] + beta[c];
    }
}

// ---------------------------------------------------------------------------
// Launcher
// ---------------------------------------------------------------------------
extern "C" void kernel_launch(void* const* d_in, const int* in_sizes, int n_in,
                              void* d_out, int out_size, void* d_ws, size_t ws_size,
                              hipStream_t stream)
{
    (void)in_sizes; (void)n_in; (void)out_size; (void)ws_size;
    const float* x      = (const float*)d_in[0];
    const float* w_qkv  = (const float*)d_in[1];
    const float* w_out  = (const float*)d_in[2];
    const float* b_out  = (const float*)d_in[3];
    const float* gamma  = (const float*)d_in[4];
    const float* beta   = (const float*)d_in[5];
    float* out = (float*)d_out;

    char* ws = (char*)d_ws;
    const size_t SZ_QKV = (size_t)NB * QKVC * HWN * sizeof(float);    // 192 MiB
    const size_t SZ_BF  = (size_t)NB * CDIM * HWN * sizeof(bf16_t);   //  32 MiB
    const size_t SZ_CTX = (size_t)NB * HEADS * 32 * 32 * sizeof(float);
    const size_t SZ_PRT = 8 * SZ_CTX;

    float*  qkv  = (float*) (ws);
    bf16_t* qT   = (bf16_t*)(ws + SZ_QKV);
    bf16_t* kS   = (bf16_t*)(ws + SZ_QKV + SZ_BF);
    bf16_t* vB   = (bf16_t*)(ws + SZ_QKV + 2 * SZ_BF);
    float*  ctx  = (float*) (ws + SZ_QKV + 3 * SZ_BF);
    float*  part = (float*) (ws + SZ_QKV + 3 * SZ_BF + SZ_CTX);
    bf16_t* attn = (bf16_t*)(ws + SZ_QKV + 3 * SZ_BF + SZ_CTX + SZ_PRT);
    float*  y0   = (float*) (ws);   // reuse qkv region (dead by kernel 8)

    // 1) QKV projection GEMM
    qkv_gemm_kernel<<<dim3(HWN / 128, QKVC / 16, NB), 128, 0, stream>>>(x, w_qkv, qkv);
    // 2) Q softmax over d -> position-major bf16
    q_softmax_kernel<<<dim3(HWN / 256, HEADS, NB), 256, 0, stream>>>(qkv, qT);
    // 3) K softmax over y -> bf16
    k_softmax_kernel<<<dim3((NB * CDIM * 64) / 256), 256, 0, stream>>>(qkv, kS);
    // 4) V -> bf16
    v_convert_kernel<<<dim3((NB * CDIM * HWN) / (256 * 8)), 256, 0, stream>>>(qkv, vB);
    // 5) context partials (K split 8x for occupancy)
    context_part_kernel<<<dim3(NB * HEADS, 8), 128, 0, stream>>>(kS, vB, part);
    // 6) deterministic partial reduction
    ctx_reduce_kernel<<<dim3((NB * HEADS * 1024) / 256), 256, 0, stream>>>(part, ctx);
    // 7) attn = ctx x qT
    apply_attn_kernel<<<dim3(HWN / 128, 2, NB * HEADS), 128, 0, stream>>>(ctx, qT, attn);
    // 8) output projection + bias
    out_proj_kernel<<<dim3(HWN / 128, CDIM / 16, NB), 128, 0, stream>>>(attn, w_out, b_out, y0);
    // 9) channel LayerNorm
    layernorm_kernel<<<dim3(HWN / 256, NB), 256, 0, stream>>>(y0, gamma, beta, out);
}
